// ComplexRNN_64123861729657
// MI455X (gfx1250) — matrix operand order repaired and checked by reading, not compiled, not run on hardware
//
#include <hip/hip_runtime.h>
#include <hip/hip_bf16.h>

// ComplexRNN for MI455X (gfx1250, wave32, WMMA).
// out[b,t,n], b<32, t<2048, n<1024.  W1: 256x1024, W2: 1024x1024 (real forms).

typedef __bf16 bf16;
typedef __attribute__((ext_vector_type(16))) __bf16 v16bf;
typedef __attribute__((ext_vector_type(8)))  __bf16 v8bf;
typedef __attribute__((ext_vector_type(8)))  float  v8f;

#define TSTEPS 2048
#define NB     32          // batch
#define N2H    1024        // 2*H
#define KIN    256         // FEAT
#define NWG    32          // persistent workgroups in scan
#define HBUF_ELEMS (NB * N2H)   // one h buffer (bf16)

// ---------------------------------------------------------------------------
// workspace init: zero sync counter + both h double-buffers (bf16 zeros)
// bytes to clear: 256 (counter pad) + 2*HBUF_ELEMS*2 = 131328 = 8208 * 16B
// ---------------------------------------------------------------------------
__global__ void ws_init_kernel(uint4* __restrict__ ws) {
  const int idx = blockIdx.x * blockDim.x + threadIdx.x;
  if (idx < 8208) ws[idx] = make_uint4(0u, 0u, 0u, 0u);
}

// ---------------------------------------------------------------------------
// Phase A: w = x @ W1 + bw  -> written into d_out (overwritten by scan later).
// Grid: (N2H/32, (B*T)/32), block 128 (4 waves), each wave one 16x16 tile.
// ---------------------------------------------------------------------------
__global__ __launch_bounds__(128) void wproj_kernel(
    const float* __restrict__ x, const float* __restrict__ Wwr,
    const float* __restrict__ Wwi, const float* __restrict__ bw,
    float* __restrict__ out)
{
  __shared__ v16bf ldsB[2][8][32];   // [ntile][kwin][lane], 16 KB
  const int wg_n0 = blockIdx.x * 32;
  const int m0g   = blockIdx.y * 32;

  // Stage W1 slice (256 x 32) into LDS in WMMA-B layout (bf16).
  for (int e = threadIdx.x; e < 512; e += 128) {
    const int ln = e & 31, kw = (e >> 5) & 7, nt = e >> 8;
    const int n  = wg_n0 + nt * 16 + (ln & 15);
    const int kb = kw * 32 + ((ln & 16) ? 16 : 0);
    v16bf v;
#pragma unroll
    for (int i = 0; i < 16; ++i) {
      const int k = kb + i;
      float val;
      if (k < 128) {
        val = (n < 512) ? Wwr[k * 512 + n] : Wwi[k * 512 + n - 512];
      } else {
        const int k2 = k - 128;
        val = (n < 512) ? -Wwi[k2 * 512 + n] : Wwr[k2 * 512 + n - 512];
      }
      v[i] = (bf16)val;
    }
    ldsB[nt][kw][ln] = v;
  }
  __syncthreads();

  const int lane  = threadIdx.x & 31;
  const int wv    = threadIdx.x >> 5;        // 0..3
  const int mrow0 = m0g + (wv >> 1) * 16;
  const int nt    = wv & 1;
  const int lm    = lane & 15;
  const int hi8   = (lane & 16) ? 16 : 0;
  const int ncol  = wg_n0 + nt * 16 + lm;
  const float bias = bw[ncol];

  v8f acc0, acc1;
#pragma unroll
  for (int i = 0; i < 8; ++i) { acc0[i] = 0.0f; acc1[i] = 0.0f; }

#pragma unroll
  for (int kw = 0; kw < 8; ++kw) {
    const float* xp = x + (size_t)(mrow0 + lm) * KIN + kw * 32 + hi8;
    const v8f x0 = *(const v8f*)(xp);
    const v8f x1 = *(const v8f*)(xp + 8);
    v16bf a;
#pragma unroll
    for (int i = 0; i < 8; ++i) { a[i] = (bf16)x0[i]; a[8 + i] = (bf16)x1[i]; }
    if (kw & 1) {
      acc1 = __builtin_amdgcn_wmma_f32_16x16x32_bf16(
          false, a, false, ldsB[nt][kw][lane], (short)0, acc1, false, false);
    } else {
      acc0 = __builtin_amdgcn_wmma_f32_16x16x32_bf16(
          false, a, false, ldsB[nt][kw][lane], (short)0, acc0, false, false);
    }
  }
  const v8f acc = acc0 + acc1;

  const int mAdd = (lane & 16) ? 8 : 0;
#pragma unroll
  for (int r = 0; r < 8; ++r)
    out[(size_t)(mrow0 + mAdd + r) * N2H + ncol] = acc[r] + bias;
}

// ---------------------------------------------------------------------------
// Phase B: persistent scan. 32 WGs x 128 threads. WG j owns columns
// [32j, 32j+32). W2 slice (1024x32) resident in LDS as bf16 (64 KB).
// h double-buffered (bf16) in d_ws; grid barrier = monotonic agent atomic.
// Dual accumulators halve the WMMA RAW chain (16 deep instead of 32).
// ---------------------------------------------------------------------------
__global__ __launch_bounds__(128) void rnn_scan_kernel(
    const float* __restrict__ Wur, const float* __restrict__ Wui,
    const float* __restrict__ bu, float* __restrict__ out,
    bf16* __restrict__ hbuf, unsigned* __restrict__ cnt)
{
  __shared__ v16bf ldsB[2][32][32];  // [ntile][kwin][lane], 64 KB
  const int wg_n0 = blockIdx.x * 32;

  // Stage W2 slice (1024 x 32) into LDS in WMMA-B layout (bf16). One time.
  for (int e = threadIdx.x; e < 2048; e += 128) {
    const int ln = e & 31, kw = (e >> 5) & 31, nt = e >> 10;
    const int n  = wg_n0 + nt * 16 + (ln & 15);
    const int kb = kw * 32 + ((ln & 16) ? 16 : 0);
    v16bf v;
#pragma unroll
    for (int i = 0; i < 16; ++i) {
      const int k = kb + i;
      float val;
      if (k < 512) {
        val = (n < 512) ? Wur[k * 512 + n] : Wui[k * 512 + n - 512];
      } else {
        const int k2 = k - 512;
        val = (n < 512) ? -Wui[k2 * 512 + n] : Wur[k2 * 512 + n - 512];
      }
      v[i] = (bf16)val;
    }
    ldsB[nt][kw][ln] = v;
  }
  __syncthreads();

  const int lane  = threadIdx.x & 31;
  const int wv    = threadIdx.x >> 5;        // 0..3
  const int mrow0 = (wv >> 1) * 16;          // batch-row tile base (0 or 16)
  const int nt    = wv & 1;
  const int lm    = lane & 15;
  const int hi8   = (lane & 16) ? 16 : 0;
  const int mAdd  = (lane & 16) ? 8 : 0;
  const int ncol  = wg_n0 + nt * 16 + lm;
  const float bias = bu[ncol];
  const int arow  = mrow0 + lm;              // A-operand row for this lane

  unsigned target = 0;
  for (int t = 0; t < TSTEPS; ++t) {
    const bf16* __restrict__ hsrc = hbuf + (size_t)(t & 1) * HBUF_ELEMS;
    bf16* __restrict__ hdst       = hbuf + (size_t)((t + 1) & 1) * HBUF_ELEMS;

    // Load w_t for this thread's 8 output elements (read-before-write,
    // same thread owns these addresses). Also prefetch w_{t+1} so the next
    // step's reads are L2/L0-resident (global_prefetch_b8, no counter cost).
    const size_t o0 = ((size_t)(mrow0 + mAdd) * TSTEPS + (size_t)t) * N2H + ncol;
    const size_t on = o0 + (size_t)((t < TSTEPS - 1) ? N2H : 0);
    float wreg[8];
#pragma unroll
    for (int r = 0; r < 8; ++r) {
      wreg[r] = out[o0 + (size_t)r * ((size_t)TSTEPS * N2H)];
      __builtin_prefetch(&out[on + (size_t)r * ((size_t)TSTEPS * N2H)], 0, 2);
    }

    v8f acc0, acc1;
#pragma unroll
    for (int i = 0; i < 8; ++i) { acc0[i] = 0.0f; acc1[i] = 0.0f; }

#pragma unroll 8
    for (int kw = 0; kw < 32; ++kw) {
      const bf16* hp = hsrc + arow * N2H + kw * 32 + hi8;
      const v8bf a0 = *(const v8bf*)(hp);
      const v8bf a1 = *(const v8bf*)(hp + 8);
      const v16bf a = __builtin_shufflevector(
          a0, a1, 0, 1, 2, 3, 4, 5, 6, 7, 8, 9, 10, 11, 12, 13, 14, 15);
      if (kw & 1) {
        acc1 = __builtin_amdgcn_wmma_f32_16x16x32_bf16(
            false, a, false, ldsB[nt][kw][lane], (short)0, acc1, false, false);
      } else {
        acc0 = __builtin_amdgcn_wmma_f32_16x16x32_bf16(
            false, a, false, ldsB[nt][kw][lane], (short)0, acc0, false, false);
      }
    }
    const v8f acc = acc0 + acc1;

#pragma unroll
    for (int r = 0; r < 8; ++r) {
      float v = wreg[r] + acc[r] + bias;
      v = v > 0.0f ? v : 0.0f;
      out[o0 + (size_t)r * ((size_t)TSTEPS * N2H)] = v;
      hdst[(mrow0 + mAdd + r) * N2H + ncol] = (bf16)v;
    }

    // ---- device-wide step barrier (monotonic counter, agent scope) ----
    __syncthreads();
    target += NWG;
    if (threadIdx.x == 0) {
      __hip_atomic_fetch_add(cnt, 1u, __ATOMIC_RELEASE, __HIP_MEMORY_SCOPE_AGENT);
      while (__hip_atomic_load(cnt, __ATOMIC_ACQUIRE, __HIP_MEMORY_SCOPE_AGENT) < target) {
        __builtin_amdgcn_s_sleep(1);
      }
    }
    __syncthreads();
    __builtin_amdgcn_fence(__ATOMIC_ACQUIRE, "agent");
  }
}

// ---------------------------------------------------------------------------
extern "C" void kernel_launch(void* const* d_in, const int* in_sizes, int n_in,
                              void* d_out, int out_size, void* d_ws, size_t ws_size,
                              hipStream_t stream) {
  const float* x   = (const float*)d_in[0];
  const float* Wwr = (const float*)d_in[1];
  const float* Wwi = (const float*)d_in[2];
  const float* bw  = (const float*)d_in[3];
  const float* Wur = (const float*)d_in[4];
  const float* Wui = (const float*)d_in[5];
  const float* bu  = (const float*)d_in[6];
  float* out = (float*)d_out;

  unsigned* cnt = (unsigned*)d_ws;
  bf16* hbuf = (bf16*)((char*)d_ws + 256);

  ws_init_kernel<<<33, 256, 0, stream>>>((uint4*)d_ws);
  wproj_kernel<<<dim3(N2H / 32, (NB * TSTEPS) / 32), 128, 0, stream>>>(
      x, Wwr, Wwi, bw, out);
  rnn_scan_kernel<<<dim3(NWG), 128, 0, stream>>>(Wur, Wui, bu, out, hbuf, cnt);
}
